// RNN_forloop_23854248362162
// MI455X (gfx1250) — matrix operand order repaired and checked by reading, not compiled, not run on hardware
//
#include <hip/hip_runtime.h>

#define I_SZ 512
#define H_SZ 1024
#define B_SZ 256
#define T_SZ 512

typedef __attribute__((ext_vector_type(16))) __bf16 v16bf;
typedef __attribute__((ext_vector_type(2)))  __bf16 v2bf;
typedef __attribute__((ext_vector_type(8)))  float  v8f;

union Frag {
  v16bf v;
  uint4 q[2];
  unsigned u[8];
};

// Pack two f32 into packed bf16x2. Prefer HW cvt if the toolchain declares it;
// fallback: round-half-up + v_perm_b32 (3 VALU ops per pair).
__device__ __forceinline__ unsigned pk_bf16x2(float a, float b) {
#if __has_builtin(__builtin_amdgcn_cvt_pk_bf16_f32)
  v2bf r = __builtin_amdgcn_cvt_pk_bf16_f32(a, b);
  return __builtin_bit_cast(unsigned, r);
#else
  unsigned ua = __builtin_bit_cast(unsigned, a) + 0x8000u;
  unsigned ub = __builtin_bit_cast(unsigned, b) + 0x8000u;
  // dest = { ub[31:24], ub[23:16], ua[31:24], ua[23:16] }
  return __builtin_amdgcn_perm(ub, ua, 0x07060302u);
#endif
}

__device__ __forceinline__ unsigned short bf16_1(float f) {
  return (unsigned short)((__builtin_bit_cast(unsigned, f) + 0x8000u) >> 16);
}

__device__ __forceinline__ v8f wmma_bf16(v16bf a, v16bf b, v8f c) {
  // D = A(16x32 bf16) * B(32x16 bf16) + C(16x16 f32)
  return __builtin_amdgcn_wmma_f32_16x16x32_bf16(false, a, false, b, (short)0, c,
                                                 false, false);
}

// tanh via v_exp_f32 + v_rcp_f32: tanh(x) = (1 - e^-2x) / (1 + e^-2x)
__device__ __forceinline__ float fast_tanh(float x) {
  x = fminf(20.0f, fmaxf(-20.0f, x));
  float e = __builtin_amdgcn_exp2f(x * -2.8853900817779268f);  // e^(-2x)
  return (1.0f - e) * __builtin_amdgcn_rcpf(1.0f + e);
}

// ---------------------------------------------------------------------------
// Prep: convert W_ih / W_hh (f32, row-major [N][K]) into bf16, pre-swizzled
// into the per-lane B-fragment order used here (contiguous k, same as A):
//   chunk ci = (n_tile * KK + kk) * 32 + lane, chunk = 16 bf16:
//   col = n_tile*16 + (lane&15), k = kk*32 + (lane>>4)*16 .. +15.
// Also bias = b_ih + b_hh.
// ---------------------------------------------------------------------------
__global__ void prep_kernel(const float* __restrict__ Wih,
                            const float* __restrict__ Whh,
                            const float* __restrict__ bih,
                            const float* __restrict__ bhh,
                            unsigned short* __restrict__ WihT,
                            unsigned short* __restrict__ WhhT,
                            float* __restrict__ bias) {
  const int idx  = blockIdx.x * blockDim.x + threadIdx.x;  // 0..65535
  const int lane = idx & 31;
  const int mr   = lane & 15;
  const int kh   = (lane >> 4) & 1;

  if (idx < 64 * 16 * 32) {  // W_ih chunks
    const int kk = (idx >> 5) & 15;
    const int tn = idx >> 9;
    const float* src = Wih + (size_t)(tn * 16 + mr) * I_SZ + kk * 32 + kh * 16;
    float4 f0 = ((const float4*)src)[0];
    float4 f1 = ((const float4*)src)[1];
    float4 f2 = ((const float4*)src)[2];
    float4 f3 = ((const float4*)src)[3];
    uint4* dst = (uint4*)(WihT + (size_t)idx * 16);
    dst[0] = make_uint4(pk_bf16x2(f0.x, f0.y), pk_bf16x2(f0.z, f0.w),
                        pk_bf16x2(f1.x, f1.y), pk_bf16x2(f1.z, f1.w));
    dst[1] = make_uint4(pk_bf16x2(f2.x, f2.y), pk_bf16x2(f2.z, f2.w),
                        pk_bf16x2(f3.x, f3.y), pk_bf16x2(f3.z, f3.w));
  }

  if (idx < 64 * 32 * 32) {  // W_hh chunks
    const int kk = (idx >> 5) & 31;
    const int tn = idx >> 10;
    const float* src = Whh + (size_t)(tn * 16 + mr) * H_SZ + kk * 32 + kh * 16;
    float4 f0 = ((const float4*)src)[0];
    float4 f1 = ((const float4*)src)[1];
    float4 f2 = ((const float4*)src)[2];
    float4 f3 = ((const float4*)src)[3];
    uint4* dst = (uint4*)(WhhT + (size_t)idx * 16);
    dst[0] = make_uint4(pk_bf16x2(f0.x, f0.y), pk_bf16x2(f0.z, f0.w),
                        pk_bf16x2(f1.x, f1.y), pk_bf16x2(f1.z, f1.w));
    dst[1] = make_uint4(pk_bf16x2(f2.x, f2.y), pk_bf16x2(f2.z, f2.w),
                        pk_bf16x2(f3.x, f3.y), pk_bf16x2(f3.z, f3.w));
  }

  if (idx < H_SZ) bias[idx] = bih[idx] + bhh[idx];
}

// ---------------------------------------------------------------------------
// x_proj GEMM (software-pipelined, depth 2):
//   out[t*B+b, j] = sum_i xs[t,b,i] * W_ih[j,i] + bias[j]
// One wave computes a 16x64 output strip (A-fragment reused across 4 N tiles).
// ---------------------------------------------------------------------------
struct AF { float4 f0, f1, f2, f3; };

__device__ __forceinline__ void xp_loadA(const float* abase, int kk, AF& af) {
  const float* ap = abase + kk * 32;
  af.f0 = ((const float4*)ap)[0];
  af.f1 = ((const float4*)ap)[1];
  af.f2 = ((const float4*)(ap + 8))[0];
  af.f3 = ((const float4*)(ap + 8))[1];
}

__device__ __forceinline__ void xp_loadB(const uint4* bb, int kk, Frag* b) {
#pragma unroll
  for (int nt = 0; nt < 4; ++nt) {
    const uint4* p = bb + nt * 1024 + kk * 64;  // tile stride 16*32*2, kk stride 32*2
    b[nt].q[0] = p[0];
    b[nt].q[1] = p[1];
  }
}

__device__ __forceinline__ void xp_pack(const AF& af, Frag& a) {
  a.u[0] = pk_bf16x2(af.f0.x, af.f0.y); a.u[1] = pk_bf16x2(af.f0.z, af.f0.w);
  a.u[2] = pk_bf16x2(af.f1.x, af.f1.y); a.u[3] = pk_bf16x2(af.f1.z, af.f1.w);
  a.u[4] = pk_bf16x2(af.f2.x, af.f2.y); a.u[5] = pk_bf16x2(af.f2.z, af.f2.w);
  a.u[6] = pk_bf16x2(af.f3.x, af.f3.y); a.u[7] = pk_bf16x2(af.f3.z, af.f3.w);
}

__global__ void __launch_bounds__(256)
xproj_kernel(const float* __restrict__ xs,
             const unsigned short* __restrict__ WihT,
             const float* __restrict__ bias,
             float* __restrict__ out) {
  const int lane = threadIdx.x & 31;
  const int half = lane >> 4;
  const int mr   = lane & 15;
  const int wgid = blockIdx.x * 8 + (threadIdx.x >> 5);
  const int tile_m = wgid >> 4;
  const int ng     = wgid & 15;

  const float* abase = xs + (size_t)(tile_m * 16 + mr) * I_SZ + half * 16;
  const uint4* bb = (const uint4*)WihT + ((size_t)(ng * 4) * 16 * 32 + lane) * 2;

  const v8f z = {0, 0, 0, 0, 0, 0, 0, 0};
  v8f acc[4] = {z, z, z, z};

  AF af0, af1;
  Frag b0[4], b1[4];
  xp_loadA(abase, 0, af0);
  xp_loadB(bb, 0, b0);

  for (int p = 0; p < 8; ++p) {  // 16 k-iters, processed in pairs
    const int k0 = 2 * p;
    xp_loadA(abase, k0 + 1, af1);
    xp_loadB(bb, k0 + 1, b1);
    {
      Frag a; xp_pack(af0, a);
#pragma unroll
      for (int nt = 0; nt < 4; ++nt) acc[nt] = wmma_bf16(a.v, b0[nt].v, acc[nt]);
    }
    if (p < 7) {
      xp_loadA(abase, k0 + 2, af0);
      xp_loadB(bb, k0 + 2, b0);
    }
    {
      Frag a; xp_pack(af1, a);
#pragma unroll
      for (int nt = 0; nt < 4; ++nt) acc[nt] = wmma_bf16(a.v, b1[nt].v, acc[nt]);
    }
  }

  const size_t orow = (size_t)(tile_m * 16 + half * 8);
#pragma unroll
  for (int nt = 0; nt < 4; ++nt) {
    const int c = (ng * 4 + nt) * 16 + mr;
    const float bv = bias[c];
#pragma unroll
    for (int r = 0; r < 8; ++r)
      out[(orow + r) * H_SZ + c] = acc[nt][r] + bv;
  }
}

// ---------------------------------------------------------------------------
// Recurrence: 16 WGs (16 batch rows each), 1024 threads = 32 waves.
// h ping-pongs in LDS (XOR-swizzled rows). Wave w owns N-tiles {2w, 2w+1}.
// Depth-2 pipelined K-loop; x_proj reads prefetched at the top of each step.
// ---------------------------------------------------------------------------
#define HROW 2048
#define HBUF (16 * HROW)  // 32 KB per h buffer; 64 KB total static LDS

__device__ __forceinline__ int lds_off(int row, int byteoff) {
  return row * HROW + (byteoff ^ (row << 4));  // XOR swizzle, 16B granularity
}

__device__ __forceinline__ void rec_load(const char* cur, const uint4* bbase,
                                         int mr, int half, int lane, int kk,
                                         Frag& a, Frag& b0, Frag& b1) {
  const int bo = (kk * 32 + half * 16) * 2;
  a.q[0] = *(const uint4*)(cur + lds_off(mr, bo));
  a.q[1] = *(const uint4*)(cur + lds_off(mr, bo + 16));
  const uint4* p0 = bbase + (size_t)(kk * 32 + lane) * 2;
  const uint4* p1 = p0 + 32 * 32 * 2;  // next N tile
  b0.q[0] = p0[0]; b0.q[1] = p0[1];
  b1.q[0] = p1[0]; b1.q[1] = p1[1];
}

__global__ void __launch_bounds__(1024)
recur_kernel(const float* __restrict__ h0,
             const unsigned short* __restrict__ WhhT,
             float* __restrict__ out) {
  __shared__ char smem[2 * HBUF];
  const int tid  = threadIdx.x;
  const int lane = tid & 31;
  const int half = lane >> 4;
  const int mr   = lane & 15;
  const int w    = tid >> 5;
  const int m0   = blockIdx.x * 16;
  const int tn0  = w * 2;

  // Load h0 (f32 -> bf16) into buffer 0
#pragma unroll
  for (int r = 0; r < 16; ++r) {
    *(unsigned short*)(smem + lds_off(r, tid * 2)) =
        bf16_1(h0[(size_t)(m0 + r) * H_SZ + tid]);
  }
  __syncthreads();

  const uint4* bbase = (const uint4*)WhhT + (size_t)tn0 * 32 * 32 * 2;
  const int c0 = tn0 * 16 + mr;
  const int c1 = c0 + 16;

  for (int t = 0; t < T_SZ; ++t) {
    const char* cur = smem + (t & 1) * HBUF;
    char* nxt = smem + ((t + 1) & 1) * HBUF;
    float* obase = out + (size_t)t * B_SZ * H_SZ + (size_t)(m0 + half * 8) * H_SZ;

    // Prefetch x_proj values: independent of the WMMA chain, hidden by K-loop
    float xp0[8], xp1[8];
#pragma unroll
    for (int r = 0; r < 8; ++r) {
      xp0[r] = obase[(size_t)r * H_SZ + c0];
      xp1[r] = obase[(size_t)r * H_SZ + c1];
    }

    v8f acc0 = {0, 0, 0, 0, 0, 0, 0, 0};
    v8f acc1 = {0, 0, 0, 0, 0, 0, 0, 0};

    Frag aA, aB, b0A, b0B, b1A, b1B;
    rec_load(cur, bbase, mr, half, lane, 0, aA, b0A, b1A);
    for (int p = 0; p < 16; ++p) {  // 32 k-iters, processed in pairs
      const int k0 = 2 * p;
      rec_load(cur, bbase, mr, half, lane, k0 + 1, aB, b0B, b1B);
      acc0 = wmma_bf16(aA.v, b0A.v, acc0);
      acc1 = wmma_bf16(aA.v, b1A.v, acc1);
      if (p < 15) rec_load(cur, bbase, mr, half, lane, k0 + 2, aA, b0A, b1A);
      acc0 = wmma_bf16(aB.v, b0B.v, acc0);
      acc1 = wmma_bf16(aB.v, b1B.v, acc1);
    }

    // Epilogue: h_new = tanh(x_proj + h W_hh^T)
#pragma unroll
    for (int r = 0; r < 8; ++r) {
      const int lr = half * 8 + r;
      float v0 = fast_tanh(acc0[r] + xp0[r]);
      float v1 = fast_tanh(acc1[r] + xp1[r]);
      obase[(size_t)r * H_SZ + c0] = v0;
      obase[(size_t)r * H_SZ + c1] = v1;
      *(unsigned short*)(nxt + lds_off(lr, c0 * 2)) = bf16_1(v0);
      *(unsigned short*)(nxt + lds_off(lr, c1 * 2)) = bf16_1(v1);
    }
    __syncthreads();
  }
}

// ---------------------------------------------------------------------------
extern "C" void kernel_launch(void* const* d_in, const int* in_sizes, int n_in,
                              void* d_out, int out_size, void* d_ws, size_t ws_size,
                              hipStream_t stream) {
  (void)in_sizes; (void)n_in; (void)out_size; (void)ws_size;
  const float* h0  = (const float*)d_in[0];
  const float* xs  = (const float*)d_in[1];
  const float* Wih = (const float*)d_in[2];
  const float* Whh = (const float*)d_in[3];
  const float* bih = (const float*)d_in[4];
  const float* bhh = (const float*)d_in[5];
  float* out = (float*)d_out;

  char* ws = (char*)d_ws;
  unsigned short* WihT = (unsigned short*)ws;                 // 1 MB bf16 fragments
  unsigned short* WhhT = (unsigned short*)(ws + (1u << 20));  // 2 MB bf16 fragments
  float* bias          = (float*)(ws + (3u << 20));           // 4 KB

  prep_kernel<<<256, 256, 0, stream>>>(Wih, Whh, bih, bhh, WihT, WhhT, bias);
  xproj_kernel<<<16384, 256, 0, stream>>>(xs, WihT, bias, out);
  recur_kernel<<<16, 1024, 0, stream>>>(h0, WhhT, out);
}
